// HiPPI_8211977470523
// MI455X (gfx1250) — matrix operand order repaired and checked by reading, not compiled, not run on hardware
//
#include <hip/hip_runtime.h>
#include <hip/hip_bf16.h>
#include <math.h>
#include <stdint.h>

typedef __attribute__((ext_vector_type(16))) __bf16 v16bf;
typedef __attribute__((ext_vector_type(8)))  __bf16 v8bf;
typedef __attribute__((ext_vector_type(8)))  float  v8f;

#define NGRAPH 64
#define MNODE  96
#define NN     (NGRAPH * MNODE)   // 6144
#define DD     128
#define SKTAU_INV 200.0f
#define SK_PAIRS 10
#define MAXIT 10

#define BM 128
#define BN 128
#define BK 32
#define LDSA 40
#define LDSB 40

#define USE_ASYNC 1

// ============================================================================
// Preconverted-operand GEMM: Cp[kc] = A @ B with A(MxK), Bt(128xK) given as
// planar bf16 hi/lo. LDS staging is contiguous 16B chunks via async-LDS DMA.
// ============================================================================
__global__ __launch_bounds__(256)
void gemm_pre(float* __restrict__ Cp,
              const __bf16* __restrict__ Ahi, const __bf16* __restrict__ Alo,
              const __bf16* __restrict__ Bthi, const __bf16* __restrict__ Btlo,
              int M, int K, const int* __restrict__ done)
{
    if (*done) return;

    __shared__ alignas(16) __bf16 As_hi[BM * LDSA];
    __shared__ alignas(16) __bf16 As_lo[BM * LDSA];
    __shared__ alignas(16) __bf16 Bt_hi[BN * LDSB];
    __shared__ alignas(16) __bf16 Bt_lo[BN * LDSB];

    const int tid  = threadIdx.x;
    const int wave = tid >> 5;
    const int lane = tid & 31;
    const int bm0  = blockIdx.x * BM;

    const int kc   = blockIdx.y;
    const int Klen = K / gridDim.y;
    const int kbeg = kc * Klen;
    const int kend = kbeg + Klen;

    const int wm = (wave & 3) * 2;
    const int wn = (wave >> 2) * 4;

    v8f acc[2][4] = {};

    for (int k0 = kbeg; k0 < kend; k0 += BK) {
        // Stage A (BM x BK, 2 planes): 512 16B-chunks/plane, 2 per thread
        #pragma unroll
        for (int i = 0; i < 2; ++i) {
            int q  = tid + 256 * i;
            int r  = q >> 2;           // 4 chunks per 32-elem row
            int c8 = (q & 3) * 8;
            const __bf16* gh = Ahi + (size_t)(bm0 + r) * K + (k0 + c8);
            const __bf16* gl = Alo + (size_t)(bm0 + r) * K + (k0 + c8);
            uint32_t dh = (uint32_t)(uintptr_t)&As_hi[r * LDSA + c8];
            uint32_t dl = (uint32_t)(uintptr_t)&As_lo[r * LDSA + c8];
#if USE_ASYNC
            asm volatile("global_load_async_to_lds_b128 %0, %1, off"
                         :: "v"(dh), "v"((uint64_t)(uintptr_t)gh) : "memory");
            asm volatile("global_load_async_to_lds_b128 %0, %1, off"
                         :: "v"(dl), "v"((uint64_t)(uintptr_t)gl) : "memory");
#else
            *(v8bf*)&As_hi[r * LDSA + c8] = *(const v8bf*)gh;
            *(v8bf*)&As_lo[r * LDSA + c8] = *(const v8bf*)gl;
#endif
        }
        // Stage Bt (128 x BK, 2 planes), already transposed in global memory
        #pragma unroll
        for (int i = 0; i < 2; ++i) {
            int q  = tid + 256 * i;
            int n  = q >> 2;
            int c8 = (q & 3) * 8;
            const __bf16* gh = Bthi + (size_t)n * K + (k0 + c8);
            const __bf16* gl = Btlo + (size_t)n * K + (k0 + c8);
            uint32_t dh = (uint32_t)(uintptr_t)&Bt_hi[n * LDSB + c8];
            uint32_t dl = (uint32_t)(uintptr_t)&Bt_lo[n * LDSB + c8];
#if USE_ASYNC
            asm volatile("global_load_async_to_lds_b128 %0, %1, off"
                         :: "v"(dh), "v"((uint64_t)(uintptr_t)gh) : "memory");
            asm volatile("global_load_async_to_lds_b128 %0, %1, off"
                         :: "v"(dl), "v"((uint64_t)(uintptr_t)gl) : "memory");
#else
            *(v8bf*)&Bt_hi[n * LDSB + c8] = *(const v8bf*)gh;
            *(v8bf*)&Bt_lo[n * LDSB + c8] = *(const v8bf*)gl;
#endif
        }
#if USE_ASYNC
        asm volatile("s_wait_asynccnt 0x0" ::: "memory");
#endif
        __syncthreads();

        const int kh = lane >> 4;
        union fr { v16bf v; v8bf h[2]; };

        fr ahi[2], alo[2];
        #pragma unroll
        for (int mt = 0; mt < 2; ++mt) {
            int arow = (wm + mt) * 16 + (lane & 15);
            ahi[mt].h[0] = *(const v8bf*)&As_hi[arow * LDSA + 8 * kh];
            ahi[mt].h[1] = *(const v8bf*)&As_hi[arow * LDSA + 16 + 8 * kh];
            alo[mt].h[0] = *(const v8bf*)&As_lo[arow * LDSA + 8 * kh];
            alo[mt].h[1] = *(const v8bf*)&As_lo[arow * LDSA + 16 + 8 * kh];
        }

        #pragma unroll
        for (int nt = 0; nt < 4; ++nt) {
            int brow = (wn + nt) * 16 + (lane & 15);
            int bk0  = 8 * kh;
            fr bhi, blo;
            bhi.h[0] = *(const v8bf*)&Bt_hi[brow * LDSB + bk0];
            bhi.h[1] = *(const v8bf*)&Bt_hi[brow * LDSB + bk0 + 16];
            blo.h[0] = *(const v8bf*)&Bt_lo[brow * LDSB + bk0];
            blo.h[1] = *(const v8bf*)&Bt_lo[brow * LDSB + bk0 + 16];

            #pragma unroll
            for (int mt = 0; mt < 2; ++mt) {
                acc[mt][nt] = __builtin_amdgcn_wmma_f32_16x16x32_bf16(
                    false, ahi[mt].v, false, bhi.v, (short)0, acc[mt][nt], false, false);
                acc[mt][nt] = __builtin_amdgcn_wmma_f32_16x16x32_bf16(
                    false, ahi[mt].v, false, blo.v, (short)0, acc[mt][nt], false, false);
                acc[mt][nt] = __builtin_amdgcn_wmma_f32_16x16x32_bf16(
                    false, alo[mt].v, false, bhi.v, (short)0, acc[mt][nt], false, false);
            }
        }
        __syncthreads();
    }

    float* Cout = Cp + (size_t)kc * M * BN;
    const int ncol = lane & 15;
    const int moff = 8 * (lane >> 4);
    #pragma unroll
    for (int mt = 0; mt < 2; ++mt) {
        const int mrow = bm0 + (wm + mt) * 16 + moff;
        #pragma unroll
        for (int nt = 0; nt < 4; ++nt) {
            #pragma unroll
            for (int r = 0; r < 8; ++r)
                Cout[(size_t)(mrow + r) * BN + (wn + nt) * 16 + ncol] = acc[mt][nt][r];
        }
    }
}

// ============================================================================
// Fallback GEMM (small-ws path): fp32 operands, on-the-fly bf16x3 split
// ============================================================================
__global__ __launch_bounds__(256)
void gemm_fly(float* __restrict__ Cp, const float* __restrict__ A,
              const float* __restrict__ B, int M, int K,
              const int* __restrict__ done)
{
    if (*done) return;
    __shared__ alignas(16) __bf16 As_hi[BM * LDSA];
    __shared__ alignas(16) __bf16 As_lo[BM * LDSA];
    __shared__ alignas(16) __bf16 Bt_hi[BN * LDSB];
    __shared__ alignas(16) __bf16 Bt_lo[BN * LDSB];

    const int tid = threadIdx.x, wave = tid >> 5, lane = tid & 31;
    const int bm0 = blockIdx.x * BM;
    const int kc = blockIdx.y;
    const int Klen = K / gridDim.y;
    const int kbeg = kc * Klen, kend = kbeg + Klen;
    const int wm = (wave & 3) * 2, wn = (wave >> 2) * 4;

    v8f acc[2][4] = {};
    for (int k0 = kbeg; k0 < kend; k0 += BK) {
        #pragma unroll
        for (int i = 0; i < (BM * BK) / 256; ++i) {
            int e = tid + 256 * i;
            int r = e >> 5, c = e & 31;
            float a = A[(size_t)(bm0 + r) * K + (k0 + c)];
            __bf16 hi = (__bf16)a;
            As_hi[r * LDSA + c] = hi;
            As_lo[r * LDSA + c] = (__bf16)(a - (float)hi);
        }
        #pragma unroll
        for (int i = 0; i < (BK * BN) / 256; ++i) {
            int e = tid + 256 * i;
            int r = e >> 7, c = e & 127;
            float b = B[(size_t)(k0 + r) * BN + c];
            __bf16 hi = (__bf16)b;
            Bt_hi[c * LDSB + r] = hi;
            Bt_lo[c * LDSB + r] = (__bf16)(b - (float)hi);
        }
        __syncthreads();
        const int kh = lane >> 4;
        union fr { v16bf v; v8bf h[2]; };
        fr ahi[2], alo[2];
        #pragma unroll
        for (int mt = 0; mt < 2; ++mt) {
            int arow = (wm + mt) * 16 + (lane & 15);
            ahi[mt].h[0] = *(const v8bf*)&As_hi[arow * LDSA + 8 * kh];
            ahi[mt].h[1] = *(const v8bf*)&As_hi[arow * LDSA + 16 + 8 * kh];
            alo[mt].h[0] = *(const v8bf*)&As_lo[arow * LDSA + 8 * kh];
            alo[mt].h[1] = *(const v8bf*)&As_lo[arow * LDSA + 16 + 8 * kh];
        }
        #pragma unroll
        for (int nt = 0; nt < 4; ++nt) {
            int brow = (wn + nt) * 16 + (lane & 15);
            int bk0 = 8 * kh;
            fr bhi, blo;
            bhi.h[0] = *(const v8bf*)&Bt_hi[brow * LDSB + bk0];
            bhi.h[1] = *(const v8bf*)&Bt_hi[brow * LDSB + bk0 + 16];
            blo.h[0] = *(const v8bf*)&Bt_lo[brow * LDSB + bk0];
            blo.h[1] = *(const v8bf*)&Bt_lo[brow * LDSB + bk0 + 16];
            #pragma unroll
            for (int mt = 0; mt < 2; ++mt) {
                acc[mt][nt] = __builtin_amdgcn_wmma_f32_16x16x32_bf16(
                    false, ahi[mt].v, false, bhi.v, (short)0, acc[mt][nt], false, false);
                acc[mt][nt] = __builtin_amdgcn_wmma_f32_16x16x32_bf16(
                    false, ahi[mt].v, false, blo.v, (short)0, acc[mt][nt], false, false);
                acc[mt][nt] = __builtin_amdgcn_wmma_f32_16x16x32_bf16(
                    false, alo[mt].v, false, bhi.v, (short)0, acc[mt][nt], false, false);
            }
        }
        __syncthreads();
    }
    float* Cout = Cp + (size_t)kc * M * BN;
    const int ncol = lane & 15;
    const int moff = 8 * (lane >> 4);
    #pragma unroll
    for (int mt = 0; mt < 2; ++mt) {
        const int mrow = bm0 + (wm + mt) * 16 + moff;
        #pragma unroll
        for (int nt = 0; nt < 4; ++nt) {
            #pragma unroll
            for (int r = 0; r < 8; ++r)
                Cout[(size_t)(mrow + r) * BN + (wn + nt) * 16 + ncol] = acc[mt][nt][r];
        }
    }
}

// ---- sum split-K partials; optionally fuse row-major bf16 hi/lo split ------
__global__ __launch_bounds__(256)
void reduce_k(float* __restrict__ C, __bf16* __restrict__ Chi,
              __bf16* __restrict__ Clo, const float* __restrict__ Cp,
              int len, int kchunks, const int* __restrict__ done)
{
    if (*done) return;
    int i = blockIdx.x * 1024 + threadIdx.x;
    #pragma unroll
    for (int j = 0; j < 4; ++j, i += 256) {
        if (i < len) {
            float s = 0.f;
            for (int sc = 0; sc < kchunks; ++sc) s += Cp[(size_t)sc * len + i];
            if (C) C[i] = s;
            if (Chi) {
                __bf16 hi = (__bf16)s;
                Chi[i] = hi;
                Clo[i] = (__bf16)(s - (float)hi);
            }
        }
    }
}

// ---- planar bf16 hi/lo split of a fp32 array (used once for W) -------------
__global__ __launch_bounds__(256)
void split_rm(__bf16* __restrict__ hiP, __bf16* __restrict__ loP,
              const float* __restrict__ X, size_t len)
{
    size_t i = (size_t)blockIdx.x * 2048 + threadIdx.x;
    #pragma unroll
    for (int j = 0; j < 8; ++j, i += 256) {
        if (i < len) {
            float v = X[i];
            __bf16 hi = (__bf16)v;
            hiP[i] = hi;
            loP[i] = (__bf16)(v - (float)hi);
        }
    }
}

// ---- Xt(128 x R) = X(R x 128)^T with fused bf16 hi/lo split ----------------
__global__ __launch_bounds__(256)
void transpose_split(__bf16* __restrict__ Xthi, __bf16* __restrict__ Xtlo,
                     const float* __restrict__ X, int R,
                     const int* __restrict__ done)
{
    if (*done) return;
    __shared__ float t[32][33];
    const int k0 = blockIdx.x * 32;   // over R
    const int i0 = blockIdx.y * 32;   // over 128
    const int lx = threadIdx.x & 31;
    const int ly = threadIdx.x >> 5;
    #pragma unroll
    for (int j = 0; j < 4; ++j)
        t[ly + 8 * j][lx] = X[(size_t)(k0 + ly + 8 * j) * DD + (i0 + lx)];
    __syncthreads();
    #pragma unroll
    for (int j = 0; j < 4; ++j) {
        float v = t[lx][ly + 8 * j];
        __bf16 hi = (__bf16)v;
        size_t o = (size_t)(i0 + ly + 8 * j) * R + (k0 + lx);
        Xthi[o] = hi;
        Xtlo[o] = (__bf16)(v - (float)hi);
    }
}

// ---- fp32 transpose (fallback path) ----------------------------------------
__global__ __launch_bounds__(256)
void transpose_u(float* __restrict__ Ut, const float* __restrict__ U,
                 const int* __restrict__ done)
{
    if (*done) return;
    __shared__ float t[32][33];
    const int k0 = blockIdx.x * 32;
    const int i0 = blockIdx.y * 32;
    const int lx = threadIdx.x & 31;
    const int ly = threadIdx.x >> 5;
    #pragma unroll
    for (int j = 0; j < 4; ++j)
        t[ly + 8 * j][lx] = U[(size_t)(k0 + ly + 8 * j) * DD + (i0 + lx)];
    __syncthreads();
    #pragma unroll
    for (int j = 0; j < 4; ++j)
        Ut[(size_t)(i0 + ly + 8 * j) * NN + (k0 + lx)] = t[lx][ly + 8 * j];
}

// ---- log-space Sinkhorn (dummy rows) + freeze + convergence partials -------
__global__ __launch_bounds__(256)
void sinkhorn_update(float* __restrict__ U, const float* __restrict__ V,
                     float* __restrict__ partials, const int* __restrict__ done)
{
    if (*done) return;
    __shared__ float ls[128 * 129];
    const int g   = blockIdx.x;
    const int tid = threadIdx.x;

    for (int i = 0; i < 64; ++i) {
        int e = tid + 256 * i;
        int r = e >> 7, c = e & 127;
        ls[r * 129 + c] = (r < MNODE)
            ? V[((size_t)g * MNODE + r) * DD + c] * SKTAU_INV : -100.0f;
    }
    __syncthreads();

    const int team = tid >> 1;
    const int sub  = tid & 1;
    for (int it = 0; it < SK_PAIRS; ++it) {
        {
            float m = -__builtin_inff();
            for (int c = sub * 64; c < sub * 64 + 64; ++c)
                m = fmaxf(m, ls[team * 129 + c]);
            m = fmaxf(m, __shfl_xor(m, 1, 32));
            float s = 0.f;
            for (int c = sub * 64; c < sub * 64 + 64; ++c)
                s += expf(ls[team * 129 + c] - m);
            s += __shfl_xor(s, 1, 32);
            float lse = m + logf(s);
            for (int c = sub * 64; c < sub * 64 + 64; ++c)
                ls[team * 129 + c] -= lse;
        }
        __syncthreads();
        {
            float m = -__builtin_inff();
            for (int r = sub * 64; r < sub * 64 + 64; ++r)
                m = fmaxf(m, ls[r * 129 + team]);
            m = fmaxf(m, __shfl_xor(m, 1, 32));
            float s = 0.f;
            for (int r = sub * 64; r < sub * 64 + 64; ++r)
                s += expf(ls[r * 129 + team] - m);
            s += __shfl_xor(s, 1, 32);
            float lse = m + logf(s);
            for (int r = sub * 64; r < sub * 64 + 64; ++r)
                ls[r * 129 + team] -= lse;
        }
        __syncthreads();
    }

    float dsq = 0.f;
    for (int i = 0; i < (MNODE * DD) / 256; ++i) {
        int e = tid + 256 * i;
        int r = e >> 7, c = e & 127;
        float cand = expf(ls[r * 129 + c]);
        size_t idx = ((size_t)g * MNODE + r) * DD + c;
        float df = cand - U[idx];
        dsq += df * df;
        U[idx] = cand;
    }
    __syncthreads();
    ls[tid] = dsq;
    __syncthreads();
    for (int s = 128; s > 0; s >>= 1) {
        if (tid < s) ls[tid] += ls[tid + s];
        __syncthreads();
    }
    if (tid == 0) partials[g] = ls[0];
}

__global__ void init_flags(int* done) { if (threadIdx.x == 0) *done = 0; }

__global__ void finalize(int* __restrict__ done, const float* __restrict__ partials)
{
    if (threadIdx.x == 0 && !*done) {
        float s = 0.f;
        for (int g = 0; g < NGRAPH; ++g) s += partials[g];
        if (sqrtf(s) < 1e-5f) *done = 1;
    }
}

extern "C" void kernel_launch(void* const* d_in, const int* in_sizes, int n_in,
                              void* d_out, int out_size, void* d_ws, size_t ws_size,
                              hipStream_t stream)
{
    (void)in_sizes; (void)n_in; (void)out_size;
    const float* W  = (const float*)d_in[0];
    const float* U0 = (const float*)d_in[1];
    float* out = (float*)d_out;

    const size_t FU = (size_t)NN * DD;          // 786432
    // ---- common fp32 section ----
    float* ws    = (float*)d_ws;
    float* U     = ws;
    float* WU    = U  + FU;
    float* V     = WU + FU;
    float* G     = V  + FU;                     // D*D
    float* parts = G  + (size_t)DD * DD;        // NGRAPH
    int*   done  = (int*)(parts + NGRAPH);      // + pad
    float* Cpart = (float*)(done + 64);         // 4*N*D
    float* tail  = Cpart + 4 * FU;              // path-specific region

    // preconverted path needs: W split 2*N*N + (Ut,WU,WUt)*2 + Gt*2 (bf16)
    const size_t bf16_elems = 2 * (size_t)NN * NN + 6 * FU + 2 * (size_t)DD * DD;
    const size_t need_pre = ((size_t)(tail - ws)) * 4 + bf16_elems * 2;

    hipMemcpyAsync(U, U0, sizeof(float) * FU, hipMemcpyDeviceToDevice, stream);
    init_flags<<<1, 32, 0, stream>>>(done);

    if (ws_size >= need_pre) {
        // ---- preconverted bf16 pipeline ----
        __bf16* Whi   = (__bf16*)tail;
        __bf16* Wlo   = Whi   + (size_t)NN * NN;
        __bf16* Uthi  = Wlo   + (size_t)NN * NN;
        __bf16* Utlo  = Uthi  + FU;
        __bf16* WUhi  = Utlo  + FU;
        __bf16* WUlo  = WUhi  + FU;
        __bf16* WUthi = WUlo  + FU;
        __bf16* WUtlo = WUthi + FU;
        __bf16* Gthi  = WUtlo + FU;
        __bf16* Gtlo  = Gthi  + (size_t)DD * DD;

        // one-time W split (bf16 planes stay L2-resident across iterations)
        split_rm<<<(int)(((size_t)NN * NN) / 2048), 256, 0, stream>>>(
            Whi, Wlo, W, (size_t)NN * NN);

        for (int it = 0; it < MAXIT; ++it) {
            transpose_split<<<dim3(NN / 32, 4), 256, 0, stream>>>(Uthi, Utlo, U, NN, done);
            // WU = W @ U   (A=W planar, B=U via transposed planes), split-K x4
            gemm_pre<<<dim3(NN / BM, 4), 256, 0, stream>>>(
                Cpart, Whi, Wlo, Uthi, Utlo, NN, NN, done);
            reduce_k<<<(NN * DD) / 1024, 256, 0, stream>>>(
                WU, WUhi, WUlo, Cpart, NN * DD, 4, done);
            transpose_split<<<dim3(NN / 32, 4), 256, 0, stream>>>(WUthi, WUtlo, WU, NN, done);
            // G = U^T @ WU (A=Ut, B=WU via WUt), split-K x48
            gemm_pre<<<dim3(DD / BM, 48), 256, 0, stream>>>(
                Cpart, Uthi, Utlo, WUthi, WUtlo, DD, NN, done);
            reduce_k<<<(DD * DD) / 1024, 256, 0, stream>>>(
                G, (__bf16*)0, (__bf16*)0, Cpart, DD * DD, 48, done);
            transpose_split<<<dim3(DD / 32, 4), 256, 0, stream>>>(Gthi, Gtlo, G, DD, done);
            // V = WU @ G   (A=WU planar, B=G via Gt), direct write
            gemm_pre<<<dim3(NN / BM, 1), 256, 0, stream>>>(
                V, WUhi, WUlo, Gthi, Gtlo, NN, DD, done);
            sinkhorn_update<<<NGRAPH, 256, 0, stream>>>(U, V, parts, done);
            finalize<<<1, 32, 0, stream>>>(done, parts);
        }
    } else {
        // ---- fallback: on-the-fly split pipeline (small workspace) ----
        float* Ut = tail;  // D*N fp32
        for (int it = 0; it < MAXIT; ++it) {
            gemm_fly<<<dim3(NN / BM, 4), 256, 0, stream>>>(Cpart, W, U, NN, NN, done);
            reduce_k<<<(NN * DD) / 1024, 256, 0, stream>>>(
                WU, (__bf16*)0, (__bf16*)0, Cpart, NN * DD, 4, done);
            transpose_u<<<dim3(NN / 32, DD / 32), 256, 0, stream>>>(Ut, U, done);
            gemm_fly<<<dim3(DD / BM, 48), 256, 0, stream>>>(Cpart, Ut, WU, DD, NN, done);
            reduce_k<<<(DD * DD) / 1024, 256, 0, stream>>>(
                G, (__bf16*)0, (__bf16*)0, Cpart, DD * DD, 48, done);
            gemm_fly<<<dim3(NN / BM, 1), 256, 0, stream>>>(V, WU, G, NN, DD, done);
            sinkhorn_update<<<NGRAPH, 256, 0, stream>>>(U, V, parts, done);
            finalize<<<1, 32, 0, stream>>>(done, parts);
        }
    }

    hipMemcpyAsync(out, U, sizeof(float) * FU, hipMemcpyDeviceToDevice, stream);
}
